// net_31421980738206
// MI455X (gfx1250) — matrix operand order repaired and checked by reading, not compiled, not run on hardware
//
#include <hip/hip_runtime.h>

#define N_NODES   100000
#define CHANNELS  32
#define K_EVECS   128
#define N_EDGES   1600000
#define CONCAT_CH 224
#define HIDDEN    256
#define OUT_CH    64

typedef __attribute__((ext_vector_type(16))) _Float16     v16h;
typedef __attribute__((ext_vector_type(8)))  float        v8f;
typedef __attribute__((ext_vector_type(4)))  unsigned int v4u;

union HFrag { _Float16 h[16]; v16h v; v4u q[2]; };

static __device__ __forceinline__ v8f wmma_f16(v16h a, v16h b, v8f c) {
  // D = A(16x32 f16) * B(32x16 f16) + C(16x16 f32)
  return __builtin_amdgcn_wmma_f32_16x16x32_f16(
      /*neg_a=*/false, a, /*neg_b=*/false, b,
      /*c_mod=*/(short)0, c, /*reuse_a=*/false, /*reuse_b=*/false);
}

// ---------------------------------------------------------------------------
// Zero coef accumulator and the SpMV scatter targets (feat cols 32..223).
// ---------------------------------------------------------------------------
__global__ void k_init(float* __restrict__ coef, float* __restrict__ feat) {
  long idx = (long)blockIdx.x * blockDim.x + threadIdx.x;
  if (idx < K_EVECS * CHANNELS) { coef[idx] = 0.f; return; }
  long r = idx - K_EVECS * CHANNELS;
  if (r < (long)N_NODES * (CONCAT_CH - CHANNELS)) {
    int n = (int)(r / (CONCAT_CH - CHANNELS));
    int c = (int)(r % (CONCAT_CH - CHANNELS));
    feat[(size_t)n * CONCAT_CH + CHANNELS + c] = 0.f;
  }
}

// ---------------------------------------------------------------------------
// Pre-transpose W1/W2 to f16, [out_channel][k] layout -> contiguous-K B frags.
// ---------------------------------------------------------------------------
__global__ void k_prep(const float* __restrict__ W1, const float* __restrict__ W2,
                       _Float16* __restrict__ W1t, _Float16* __restrict__ W2t) {
  int i = blockIdx.x * blockDim.x + threadIdx.x;
  if (i < CONCAT_CH * HIDDEN) {
    int c = i / CONCAT_CH, k = i % CONCAT_CH;
    W1t[i] = (_Float16)W1[k * HIDDEN + c];
  } else {
    int j = i - CONCAT_CH * HIDDEN;
    if (j < HIDDEN * OUT_CH) {
      int c = j / HIDDEN, k = j % HIDDEN;
      W2t[j] = (_Float16)W2[k * OUT_CH + c];
    }
  }
}

// ---------------------------------------------------------------------------
// coef[k,c] = sum_n evecs[n,k] * x[n,c]  (split-N reduction, LDS tiles,
// register accumulators, global f32 atomics at the end). Prefetch next batch.
// ---------------------------------------------------------------------------
__global__ void k_coef(const float* __restrict__ x, const float* __restrict__ ev,
                       float* __restrict__ coef) {
  __shared__ float evs[16][K_EVECS];
  __shared__ float xs[16][CHANNELS];
  int t  = threadIdx.x;
  int c  = t & 31;   // owned channel
  int kb = t >> 5;   // k % 8 group
  float acc[16];
#pragma unroll
  for (int i = 0; i < 16; ++i) acc[i] = 0.f;
  int start = blockIdx.x * 256;
  int end   = start + 256; if (end > N_NODES) end = N_NODES;
  for (int nb = start; nb < end; nb += 16) {
    int cnt = end - nb; if (cnt > 16) cnt = 16;
    for (int i = t; i < 16 * K_EVECS; i += 256) {
      int b = i >> 7, k = i & (K_EVECS - 1);
      evs[b][k] = (b < cnt) ? ev[(size_t)(nb + b) * K_EVECS + k] : 0.f;
    }
    for (int i = t; i < 16 * CHANNELS; i += 256) {
      int b = i >> 5, cc = i & (CHANNELS - 1);
      xs[b][cc] = (b < cnt) ? x[(size_t)(nb + b) * CHANNELS + cc] : 0.f;
    }
    // prefetch next 16-row evecs batch while this one computes
    if (nb + 16 + (t >> 4) < end)
      __builtin_prefetch(&ev[(size_t)(nb + 16 + (t >> 4)) * K_EVECS + ((t & 15) << 3)], 0, 0);
    __syncthreads();
#pragma unroll 4
    for (int b = 0; b < 16; ++b) {
      float xv = xs[b][c];
#pragma unroll
      for (int i = 0; i < 16; ++i) acc[i] = fmaf(evs[b][kb + 8 * i], xv, acc[i]);
    }
    __syncthreads();
  }
#pragma unroll
  for (int i = 0; i < 16; ++i)
    atomicAdd(&coef[(kb + 8 * i) * CHANNELS + c], acc[i]);
}

// ---------------------------------------------------------------------------
// coefS[k,c] = exp(-softplus(tau) * evals[k]) * coef[k,c]
// ---------------------------------------------------------------------------
__global__ void k_scale(const float* __restrict__ tau, const float* __restrict__ evals,
                        const float* __restrict__ coef, float* __restrict__ coefS) {
  int k = threadIdx.x;  // 128 threads
  float tv = tau[0];
  float t  = (tv > 20.f) ? tv : log1pf(expf(tv));
  float g  = expf(-t * evals[k]);
#pragma unroll
  for (int c = 0; c < CHANNELS; ++c)
    coefS[k * CHANNELS + c] = g * coef[k * CHANNELS + c];
}

// ---------------------------------------------------------------------------
// feat[:,0:32] = rownorm(evecs @ coefS) via v_wmma_f32_16x16x32_f16.
// coefS staged TRANSPOSED [c][k] in LDS so B fragments are 2x ds_load_b128.
// ---------------------------------------------------------------------------
__global__ void k_diffuse(const float* __restrict__ evecs,
                          const float* __restrict__ coefS,
                          float* __restrict__ feat) {
  __shared__ _Float16 cst[CHANNELS * K_EVECS];  // [c][k] f16
  int tid = threadIdx.x;
  for (int i = tid; i < CHANNELS * K_EVECS; i += 256) {
    int c = i >> 7, k = i & (K_EVECS - 1);
    cst[i] = (_Float16)coefS[k * CHANNELS + c];
  }
  __syncthreads();
  int wave = tid >> 5, lane = tid & 31;
  int m = lane & 15, khalf = lane >> 4, col = lane & 15;
  int rowBase = blockIdx.x * 128 + wave * 16;
  int nrow = rowBase + m; if (nrow > N_NODES - 1) nrow = N_NODES - 1;  // clamp: EXEC stays full
  const float* er = evecs + (size_t)nrow * K_EVECS;
  v8f acc0 = {}, acc1 = {};
  for (int kt = 0; kt < 4; ++kt) {
    int k0a = kt * 32 + khalf * 16;  // A: contiguous 16 K per half-wave
    HFrag a;
#pragma unroll
    for (int j = 0; j < 8; ++j) {
      a.h[j]     = (_Float16)er[k0a + j];
      a.h[8 + j] = (_Float16)er[k0a + 8 + j];
    }
    int k0b = kt * 32 + khalf * 16;  // B: contiguous 16 K per half-wave
    HFrag b0, b1;
    const v4u* p0 = (const v4u*)(cst + col * K_EVECS + k0b);
    const v4u* p1 = (const v4u*)(cst + (16 + col) * K_EVECS + k0b);
    b0.q[0] = p0[0]; b0.q[1] = p0[1];
    b1.q[0] = p1[0]; b1.q[1] = p1[1];
    acc0 = wmma_f16(a.v, b0.v, acc0);
    acc1 = wmma_f16(a.v, b1.v, acc1);
  }
#pragma unroll
  for (int r = 0; r < 8; ++r) {
    float v0 = acc0[r], v1 = acc1[r];
    float s = v0 * v0 + v1 * v1;
    s += __shfl_xor(s, 1);   // reduce across the 16-lane half (cols 0..31)
    s += __shfl_xor(s, 2);
    s += __shfl_xor(s, 4);
    s += __shfl_xor(s, 8);
    float nrm = sqrtf(s);
    nrm = nrm > 1e-12f ? nrm : 1e-12f;
    float inv = 1.0f / nrm;
    int row = rowBase + r + khalf * 8;
    if (row < N_NODES) {
      feat[(size_t)row * CONCAT_CH + col]      = v0 * inv;
      feat[(size_t)row * CONCAT_CH + 16 + col] = v1 * inv;
    }
  }
}

// ---------------------------------------------------------------------------
// COO SpMV: one thread per edge; float4 gather of x[col], scale, f32 atomic
// scatter-add into out[row]. in/out live inside the 224-wide feat buffer.
// ---------------------------------------------------------------------------
__global__ void k_spmv(const int* __restrict__ rows, const int* __restrict__ cols,
                       const float* __restrict__ vals,
                       const float* __restrict__ xin, float* __restrict__ outp,
                       int C) {
  int e = blockIdx.x * blockDim.x + threadIdx.x;
  if (e >= N_EDGES) return;
  int r = rows[e], cc = cols[e];
  float v = vals[e];
  const float4* src = (const float4*)(xin + (size_t)cc * CONCAT_CH);
  float* dst = outp + (size_t)r * CONCAT_CH;
  int q = C >> 2;
  for (int c4 = 0; c4 < q; ++c4) {
    float4 mv = src[c4];
    atomicAdd(dst + c4 * 4 + 0, v * mv.x);
    atomicAdd(dst + c4 * 4 + 1, v * mv.y);
    atomicAdd(dst + c4 * 4 + 2, v * mv.z);
    atomicAdd(dst + c4 * 4 + 3, v * mv.w);
  }
}

// ---------------------------------------------------------------------------
// Fused MLP: emb = relu(feat@W1 + b1) @ W2 + b2, all matmuls on WMMA f16.
// LDS: W1t (112KB f16) + W2t (32KB f16) + per-wave hidden staging (64KB)
//    = 208KB of the 320KB WGP budget. All fragments load as ds_load_b128.
// ---------------------------------------------------------------------------
__global__ void k_mlp(const float* __restrict__ feat,
                      const _Float16* __restrict__ W1t,
                      const _Float16* __restrict__ W2t,
                      const float* __restrict__ b1,
                      const float* __restrict__ b2,
                      float* __restrict__ emb) {
  extern __shared__ _Float16 smem[];
  _Float16* w1   = smem;                              // [HIDDEN][CONCAT_CH]
  _Float16* w2   = w1 + HIDDEN * CONCAT_CH;           // [OUT_CH][HIDDEN]
  _Float16* hidb = w2 + OUT_CH * HIDDEN;              // [8 waves][16][HIDDEN]
  int tid = threadIdx.x;
  {  // vectorized weight staging: b128 global loads -> b128 LDS stores
    const v4u* s1 = (const v4u*)W1t; v4u* d1 = (v4u*)w1;
    for (int i = tid; i < (HIDDEN * CONCAT_CH) / 8; i += 256) d1[i] = s1[i];
    const v4u* s2 = (const v4u*)W2t; v4u* d2 = (v4u*)w2;
    for (int i = tid; i < (OUT_CH * HIDDEN) / 8; i += 256) d2[i] = s2[i];
  }
  __syncthreads();

  int wave = tid >> 5, lane = tid & 31;
  int m = lane & 15, khalf = lane >> 4, col = lane & 31 & 15;
  int rowBase = blockIdx.x * 128 + wave * 16;
  int nrow = rowBase + m; if (nrow > N_NODES - 1) nrow = N_NODES - 1;
  const float* fr = feat + (size_t)nrow * CONCAT_CH;

  // ---- layer 1: [16 x 224] @ [224 x 256] -> acc[16] tiles of 16x16 f32 ----
  v8f acc[16] = {};
  for (int kt = 0; kt < 7; ++kt) {                 // K = 224 = 7 * 32
    int k0a = kt * 32 + khalf * 16;
    HFrag a;
#pragma unroll
    for (int j = 0; j < 8; ++j) {
      a.h[j]     = (_Float16)fr[k0a + j];
      a.h[8 + j] = (_Float16)fr[k0a + 8 + j];
    }
    int k0b = kt * 32 + khalf * 16;
#pragma unroll
    for (int nt = 0; nt < 16; ++nt) {              // N = 256 = 16 tiles
      const v4u* wp = (const v4u*)(w1 + (nt * 16 + col) * CONCAT_CH + k0b);
      HFrag b;
      b.q[0] = wp[0]; b.q[1] = wp[1];
      acc[nt] = wmma_f16(a.v, b.v, acc[nt]);
    }
  }

  // ---- bias + relu, stage hidden (f16) to LDS in A-fragment-friendly layout
  _Float16* hrow = hidb + wave * 16 * HIDDEN;
#pragma unroll
  for (int nt = 0; nt < 16; ++nt) {
    int cIdx = nt * 16 + col;
    float bias = b1[cIdx];
#pragma unroll
    for (int r = 0; r < 8; ++r) {
      float hv = acc[nt][r] + bias;
      hv = hv > 0.f ? hv : 0.f;
      hrow[(r + khalf * 8) * HIDDEN + cIdx] = (_Float16)hv;
    }
  }
  __syncthreads();

  // ---- layer 2: [16 x 256] @ [256 x 64] ----
  v8f acc2[4] = {};
  for (int kt = 0; kt < 8; ++kt) {                 // K = 256 = 8 * 32
    int k0a = kt * 32 + khalf * 16;
    const _Float16* hm = hrow + m * HIDDEN;
    HFrag a;
    const v4u* pa = (const v4u*)(hm + k0a);        // 16B-aligned, +16B for hi 8
    a.q[0] = pa[0]; a.q[1] = pa[1];
    int k0b = kt * 32 + khalf * 16;
#pragma unroll
    for (int nt = 0; nt < 4; ++nt) {               // N = 64 = 4 tiles
      const v4u* wp = (const v4u*)(w2 + (nt * 16 + col) * HIDDEN + k0b);
      HFrag b;
      b.q[0] = wp[0]; b.q[1] = wp[1];
      acc2[nt] = wmma_f16(a.v, b.v, acc2[nt]);
    }
  }

#pragma unroll
  for (int nt = 0; nt < 4; ++nt) {
    int cIdx = nt * 16 + col;
    float bias = b2[cIdx];
#pragma unroll
    for (int r = 0; r < 8; ++r) {
      int row = rowBase + r + khalf * 8;
      if (row < N_NODES) emb[(size_t)row * OUT_CH + cIdx] = acc2[nt][r] + bias;
    }
  }
}

// ---------------------------------------------------------------------------
extern "C" void kernel_launch(void* const* d_in, const int* in_sizes, int n_in,
                              void* d_out, int out_size, void* d_ws, size_t ws_size,
                              hipStream_t stream) {
  (void)in_sizes; (void)n_in; (void)out_size; (void)ws_size;
  const float* x     = (const float*)d_in[0];
  const float* evals = (const float*)d_in[1];
  const float* evecs = (const float*)d_in[2];
  const float* tau   = (const float*)d_in[3];
  const float* vals  = (const float*)d_in[4];
  const float* W1    = (const float*)d_in[5];
  const float* b1    = (const float*)d_in[6];
  const float* W2    = (const float*)d_in[7];
  const float* b2    = (const float*)d_in[8];
  const int*   rows  = (const int*)d_in[9];
  const int*   cols  = (const int*)d_in[10];
  float* out = (float*)d_out;

  // workspace layout
  char* ws = (char*)d_ws;
  float*    coef  = (float*)ws;                                   // 16 KB
  float*    coefS = (float*)(ws + 16384);                         // 16 KB
  float*    feat  = (float*)(ws + 32768);                         // N*224 f32
  size_t featBytes = (size_t)N_NODES * CONCAT_CH * sizeof(float);
  _Float16* W1t = (_Float16*)(ws + 32768 + featBytes);
  _Float16* W2t = W1t + CONCAT_CH * HIDDEN;

  // 1. zero coef + scatter targets
  long nZero = (long)K_EVECS * CHANNELS + (long)N_NODES * (CONCAT_CH - CHANNELS);
  k_init<<<(unsigned)((nZero + 255) / 256), 256, 0, stream>>>(coef, feat);

  // 2. weights -> transposed f16
  int nPrep = CONCAT_CH * HIDDEN + HIDDEN * OUT_CH;
  k_prep<<<(nPrep + 255) / 256, 256, 0, stream>>>(W1, W2, W1t, W2t);

  // 3. coef = evecs^T @ x
  k_coef<<<(N_NODES + 255) / 256, 256, 0, stream>>>(x, evecs, coef);

  // 4. spectral scaling
  k_scale<<<1, K_EVECS, 0, stream>>>(tau, evals, coef, coefS);

  // 5. diffusion + row-normalize -> feat[:, 0:32]
  k_diffuse<<<(N_NODES + 127) / 128, 256, 0, stream>>>(evecs, coefS, feat);

  // 6. order-1 gradients (input cols 0..31, outputs cols 32..63 / 64..95)
  int eb = (N_EDGES + 255) / 256;
  k_spmv<<<eb, 256, 0, stream>>>(rows,           cols,           vals,
                                 feat + 0,  feat + 32,  CHANNELS);
  k_spmv<<<eb, 256, 0, stream>>>(rows + N_EDGES, cols + N_EDGES, vals + N_EDGES,
                                 feat + 0,  feat + 64,  CHANNELS);
  // 7. order-2 gradients (input cols 32..95, outputs cols 96..159 / 160..223)
  k_spmv<<<eb, 256, 0, stream>>>(rows,           cols,           vals,
                                 feat + 32, feat + 96,  2 * CHANNELS);
  k_spmv<<<eb, 256, 0, stream>>>(rows + N_EDGES, cols + N_EDGES, vals + N_EDGES,
                                 feat + 32, feat + 160, 2 * CHANNELS);

  // 8. fused MLP on WMMA f16
  size_t mlpLds = (size_t)(HIDDEN * CONCAT_CH + OUT_CH * HIDDEN + 8 * 16 * HIDDEN)
                  * sizeof(_Float16);
  k_mlp<<<(N_NODES + 127) / 128, 256, mlpLds, stream>>>(feat, W1t, W2t, b1, b2, out);
}
